// SwitchGate_79156247265932
// MI455X (gfx1250) — matrix operand order repaired and checked by reading, not compiled, not run on hardware
//
#include <hip/hip_runtime.h>
#include <hip/hip_bf16.h>

// Problem dims (fixed by the reference)
#define BDIM 16
#define SDIM 2048
#define DDIM 1024
#define EDIM 64
#define NTOK (BDIM * SDIM)   // 32768 tokens
#define KSTEPS (DDIM / 32)   // 32 K-iterations of depth 32

typedef __attribute__((ext_vector_type(16))) __bf16    v16bf;
typedef __attribute__((ext_vector_type(8)))  float     v8f;
typedef __attribute__((ext_vector_type(4)))  float     v4f;
typedef __attribute__((ext_vector_type(8)))  unsigned  v8u;

// Pack two f32 -> two bf16 (round-to-nearest, ties-up) in one u32:
// 2x v_add_u32 + 1x v_perm_b32.  perm sel 0x07060302 picks
// {src0.b3, src0.b2, src1.b3, src1.b2} = {hi16(f1), hi16(f0)}.
static __device__ __forceinline__ unsigned pk_bf16(float f0, float f1) {
    unsigned u0 = __builtin_bit_cast(unsigned, f0) + 0x8000u;
    unsigned u1 = __builtin_bit_cast(unsigned, f1) + 0x8000u;
    return __builtin_amdgcn_perm(u1, u0, 0x07060302u);
}

// ---------------------------------------------------------------------------
// Phase 0: Wg f32 [E=64][D=1024] -> bf16, pre-swizzled into the WMMA B-matrix
// lane layout (16-bit B 32x16: lanes 0-15 hold K=kb+0..7,16..23 for N=lane&15,
// lanes 16-31 hold K=kb+8..15,24..31). Flat layout: [ks][etile][lane][j].
// ---------------------------------------------------------------------------
__global__ __launch_bounds__(256) void wg_swizzle_kernel(
    const float* __restrict__ Wg, unsigned short* __restrict__ Wswz) {
    int t    = blockIdx.x * blockDim.x + threadIdx.x;  // 0..65535
    int j    = t & 15;
    int lane = (t >> 4) & 31;
    int et   = (t >> 9) & 3;
    int ks   = t >> 11;
    int e = et * 16 + (lane & 15);
    int k = ks * 32 + ((j >> 3) << 4) + ((lane >> 4) << 3) + (j & 7);
    // RNE here (one-time cost)
    unsigned u = __builtin_bit_cast(unsigned, Wg[e * DDIM + k]);
    u += 0x7FFFu + ((u >> 16) & 1u);
    Wswz[t] = (unsigned short)(u >> 16);
}

// ---------------------------------------------------------------------------
// Phase 1: logits GEMM (bf16 WMMA, f32 accum) + bias + softmax + top-2 mask.
// 1 wave = 16 tokens x 64 experts, double-buffered over K. 8 waves / block.
// Writes masked gate probabilities into `masked` = d_out layout [B,S,E].
// ---------------------------------------------------------------------------
__global__ __launch_bounds__(256) void gate_gemm_kernel(
    const float* __restrict__ X, const unsigned short* __restrict__ Wswz,
    const float* __restrict__ bg, float* __restrict__ masked) {
    __shared__ float lds[8][16][EDIM];   // per-wave 16x64 logit tile (32 KB)

    const int tid  = threadIdx.x;
    const int wave = tid >> 5;
    const int l    = tid & 31;
    const int half = l >> 4;     // 0: K 0..7 / 16..23 ; 1: K 8..15 / 24..31
    const int m    = l & 15;     // token row within tile
    const int token0 = (blockIdx.x * 8 + wave) * 16;

    const float* xrow = X + (size_t)(token0 + m) * DDIM;

    v8f acc[4];
#pragma unroll
    for (int et = 0; et < 4; ++et)
#pragma unroll
        for (int r = 0; r < 8; ++r) acc[et][r] = 0.0f;

    // Double-buffered fragments: A raw f32 runs + 4 swizzled B fragments.
    v4f   xv[2][4];
    v16bf bv[2][4];

    auto load_step = [&](int ks, int buf) {
        const int kb = ks * 32 + half * 8;            // A: two 32B runs/lane
        xv[buf][0] = *(const v4f*)(xrow + kb);
        xv[buf][1] = *(const v4f*)(xrow + kb + 4);
        xv[buf][2] = *(const v4f*)(xrow + kb + 16);
        xv[buf][3] = *(const v4f*)(xrow + kb + 20);
        const unsigned short* wk = Wswz + ((size_t)ks * 4 * 32 + l) * 16;
        bv[buf][0] = *(const v16bf*)(wk);             // coalesced 32B/lane
        bv[buf][1] = *(const v16bf*)(wk + 32 * 16);
        bv[buf][2] = *(const v16bf*)(wk + 2 * 32 * 16);
        bv[buf][3] = *(const v16bf*)(wk + 3 * 32 * 16);
    };

    load_step(0, 0);

#pragma unroll 2
    for (int ks = 0; ks < KSTEPS; ++ks) {
        const int cur = ks & 1;
        const int nxt = cur ^ 1;
        if (ks + 1 < KSTEPS) load_step(ks + 1, nxt);  // prefetch next K-step

        // --- A fragment: 16-bit A 16x32 layout (ISA 7.12.2) ---
        // 8 packed u32 lanes = 16 bf16 elements (pair order preserved).
        v8u au;
        au[0] = pk_bf16(xv[cur][0][0], xv[cur][0][1]);
        au[1] = pk_bf16(xv[cur][0][2], xv[cur][0][3]);
        au[2] = pk_bf16(xv[cur][1][0], xv[cur][1][1]);
        au[3] = pk_bf16(xv[cur][1][2], xv[cur][1][3]);
        au[4] = pk_bf16(xv[cur][2][0], xv[cur][2][1]);
        au[5] = pk_bf16(xv[cur][2][2], xv[cur][2][3]);
        au[6] = pk_bf16(xv[cur][3][0], xv[cur][3][1]);
        au[7] = pk_bf16(xv[cur][3][2], xv[cur][3][3]);
        v16bf a = __builtin_bit_cast(v16bf, au);

#pragma unroll
        for (int et = 0; et < 4; ++et) {
            acc[et] = __builtin_amdgcn_wmma_f32_16x16x32_bf16(
                false, a, false, bv[cur][et], (short)0, acc[et], false, false);
        }
    }

    // --- spill logits (+bias) to LDS: C/D layout VGPR r -> M = r + 8*half ---
#pragma unroll
    for (int et = 0; et < 4; ++et) {
        float bge = bg[et * 16 + m];
#pragma unroll
        for (int r = 0; r < 8; ++r)
            lds[wave][r + half * 8][et * 16 + m] = acc[et][r] + bge;
    }
    // Same-wave LDS RAW: make the spill visible before re-reading.
    asm volatile("s_wait_dscnt 0" ::: "memory");

    // --- per-row softmax + top-2 (lanes 0..15, one token row each) ---
    if (l < 16) {
        const v4f* r4 = (const v4f*)&lds[wave][l][0];
        float b1 = -3.402823466e38f, b2 = -3.402823466e38f;
        int   i1 = -1, i2 = -1;
#pragma unroll
        for (int c = 0; c < 16; ++c) {
            v4f t4 = r4[c];
#pragma unroll
            for (int q = 0; q < 4; ++q) {
                float t = t4[q];
                int   e = c * 4 + q;
                if (t > b1)      { b2 = b1; i2 = i1; b1 = t; i1 = e; }
                else if (t > b2) { b2 = t; i2 = e; }
            }
        }
        float ssum = 0.0f;
#pragma unroll
        for (int c = 0; c < 16; ++c) {
            v4f t4 = r4[c];
#pragma unroll
            for (int q = 0; q < 4; ++q) ssum += __expf(t4[q] - b1);
        }
        const float inv = 1.0f / ssum;
        const float p1  = inv;                     // exp(0)/sum
        const float p2  = __expf(b2 - b1) * inv;

        v4f* orow = (v4f*)(masked + (size_t)(token0 + l) * EDIM);
#pragma unroll
        for (int c = 0; c < 16; ++c) {
            v4f w;
#pragma unroll
            for (int q = 0; q < 4; ++q) {
                int e = c * 4 + q;
                w[q] = (e == i1) ? p1 : ((e == i2) ? p2 : 0.0f);
            }
            orow[c] = w;
        }
    }
}

// ---------------------------------------------------------------------------
// Phase 2: per-(s,e) batch-axis normalization, in place on d_out.
// out[b,s,e] = masked[b,s,e] / (sum_b masked[b,s,e] + eps) * capacity(=16)
// One block per s; tid -> (q = tid/64, e = tid%64), each thread owns 4 b's.
// ---------------------------------------------------------------------------
__global__ __launch_bounds__(256) void norm_kernel(float* __restrict__ out) {
    __shared__ float part[4][EDIM];
    const int s   = blockIdx.x;
    const int tid = threadIdx.x;
    const int e   = tid & 63;
    const int q   = tid >> 6;

    float vals[4];
    float sum = 0.0f;
#pragma unroll
    for (int i = 0; i < 4; ++i) {
        int b = q * 4 + i;
        vals[i] = out[((size_t)b * SDIM + s) * EDIM + e];
        sum += vals[i];
    }
    part[q][e] = sum;
    __syncthreads();
    const float denom = part[0][e] + part[1][e] + part[2][e] + part[3][e] + 1e-6f;
    const float scale = 16.0f / denom;   // capacity = int(1.0 * B) = 16
#pragma unroll
    for (int i = 0; i < 4; ++i) {
        int b = q * 4 + i;
        out[((size_t)b * SDIM + s) * EDIM + e] = vals[i] * scale;
    }
}

// ---------------------------------------------------------------------------
extern "C" void kernel_launch(void* const* d_in, const int* in_sizes, int n_in,
                              void* d_out, int out_size, void* d_ws, size_t ws_size,
                              hipStream_t stream) {
    const float* X  = (const float*)d_in[0];   // [16,2048,1024] f32
    const float* Wg = (const float*)d_in[1];   // [64,1024] f32
    const float* bg = (const float*)d_in[2];   // [64] f32
    float* out = (float*)d_out;                // [16,2048,64] f32

    unsigned short* Wswz = (unsigned short*)d_ws;  // 128 KB bf16 swizzled Wg

    wg_swizzle_kernel<<<dim3(EDIM * DDIM / 256), dim3(256), 0, stream>>>(Wg, Wswz);
    gate_gemm_kernel<<<dim3(NTOK / (16 * 8)), dim3(256), 0, stream>>>(X, Wswz, bg, out);
    norm_kernel<<<dim3(SDIM), dim3(256), 0, stream>>>(out);
}
